// DenseGATConv_41515153883483
// MI455X (gfx1250) — matrix operand order repaired and checked by reading, not compiled, not run on hardware
//
#include <hip/hip_runtime.h>
#include <hip/hip_bf16.h>
#include <cstdint>

// ---------------------------------------------------------------------------
// DenseGATConv on MI455X (gfx1250, wave32, WMMA)
//   K1: feat = x@W        (v_wmma_f32_16x16x4_f32), also emits feat16^T
//   K2: per-(b,n,h) attention scalars -> p=e^{asrc}, p2=e^{.2asrc}, q, q2
//   K3: column denominators S[b,j,h] = sum_i adj*sel(p_i q_j)  -> invS
//   K4: o[b,i,h,c] = sum_j alpha*feat via v_wmma_f32_16x16x32_f16,
//       32-row i-tiles, double-buffered async adj staging (ASYNCcnt).
// exp(leaky(s)) trick:  s>0 <=> p_i*q_j>1 ;  value = p_i*q_j or p2_i*q2_j
// ---------------------------------------------------------------------------

#define BB    4
#define NN    2048
#define FIN   128
#define HH    4
#define CC    32
#define HC    128   // H*C
#define ITILE 32    // i-rows per block in K4
#define LPAD  36    // padded LDS row stride (floats): b128-aligned, bank-safe

typedef __attribute__((ext_vector_type(16))) _Float16 v16h;
typedef __attribute__((ext_vector_type(8)))  float    v8f;
typedef __attribute__((ext_vector_type(4)))  float    v4f;
typedef __attribute__((ext_vector_type(2)))  float    v2f;
typedef __attribute__((ext_vector_type(4)))  int      v4i;

#if defined(__has_builtin)
#if __has_builtin(__builtin_amdgcn_wmma_f32_16x16x4_f32)
#define HAVE_WMMA_F32X4 1
#endif
#if __has_builtin(__builtin_amdgcn_global_load_async_to_lds_b128) && \
    __has_builtin(__builtin_amdgcn_s_wait_asynccnt)
#define HAVE_ASYNC_LDS 1
#endif
#endif

#if defined(HAVE_ASYNC_LDS)
__device__ __forceinline__ void async_cp_b128(const float* gsrc, float* ldst)
{
    // Builtin params are int4 pointers in AS1 (global) / AS3 (LDS); reinterpret
    // through uintptr_t (low 32 bits of a generic LDS pointer are the LDS
    // byte offset per the ISA flat-addressing rules).
    __builtin_amdgcn_global_load_async_to_lds_b128(
        (__attribute__((address_space(1))) v4i*)(uintptr_t)gsrc,
        (__attribute__((address_space(3))) v4i*)(uintptr_t)ldst,
        0, 0);
}
#endif

// ---------------------------------------------------------------- K1: x @ W
// grid = B*N/16 blocks of 256 (8 waves); wave w -> output cols [16w,16w+16)
__global__ __launch_bounds__(256) void k_project(
    const float* __restrict__ x, const float* __restrict__ W,
    float* __restrict__ feat, _Float16* __restrict__ featT16)
{
    const int gid  = blockIdx.x;
    const int b    = gid / (NN / 16);
    const int i0   = (gid % (NN / 16)) * 16;
    const int wave = threadIdx.x >> 5;
    const int lane = threadIdx.x & 31;
    const int M    = lane & 15;           // A-operand row for this lane
    const int hi   = lane >> 4;           // lane group: K sub-slot
    const int col  = wave * 16 + M;       // B/C-operand column for this lane

    const float* __restrict__ arow = x + ((size_t)b * NN + i0 + M) * FIN;
    v8f acc = {};

#if defined(HAVE_WMMA_F32X4)
    // A 16x4 f32: lanes0-15 hold K0,K1 ; lanes16-31 hold K2,K3 (ISA 7.12.2)
    // B 4x16 f32: VGPR0 lanes0-15 K0 / lanes16-31 K2 ; VGPR1 K1 / K3
    #pragma unroll 4
    for (int k0 = 0; k0 < FIN; k0 += 4) {
        v2f a, bb;
        a.x  = arow[k0 + hi * 2 + 0];
        a.y  = arow[k0 + hi * 2 + 1];
        bb.x = W[(size_t)(k0 + hi * 2 + 0) * HC + col];
        bb.y = W[(size_t)(k0 + hi * 2 + 1) * HC + col];
        acc = __builtin_amdgcn_wmma_f32_16x16x4_f32(
            false, a, false, bb, (short)0, acc, false, false);
    }
#else
    #pragma unroll
    for (int r = 0; r < 8; ++r) {
        const float* __restrict__ xr = x + ((size_t)b * NN + i0 + r + hi * 8) * FIN;
        float s = 0.f;
        for (int k = 0; k < FIN; ++k) s += xr[k] * W[(size_t)k * HC + col];
        acc[r] = s;
    }
#endif

    // C/D 16x16 f32: VGPR r -> M=r (lanes0-15) / M=r+8 (lanes16-31), N=lane&15
    #pragma unroll
    for (int r = 0; r < 8; ++r) {
        const int row = i0 + r + hi * 8;
        feat[((size_t)b * NN + row) * HC + col]        = acc[r];
        featT16[((size_t)b * HC + col) * NN + row]     = (_Float16)acc[r];
    }
}

// -------------------------------------------- K2: attention scalars + exps
// one wave per (b,n); lane = channel within head
__global__ __launch_bounds__(256) void k_attn(
    const float* __restrict__ feat,
    const float* __restrict__ att_src, const float* __restrict__ att_dst,
    float* __restrict__ p, float* __restrict__ p2,
    float* __restrict__ q, float* __restrict__ q2)
{
    const int wave = threadIdx.x >> 5;
    const int lane = threadIdx.x & 31;
    const int row  = blockIdx.x * 8 + wave;           // 0 .. B*N-1
    const int b    = row / NN;
    const int n    = row % NN;

    const float* __restrict__ fr = feat + ((size_t)b * NN + n) * HC;
    float v0 = fr[0 * 32 + lane], v1 = fr[1 * 32 + lane];
    float v2 = fr[2 * 32 + lane], v3 = fr[3 * 32 + lane];

    float s0 = v0 * att_src[0 * 32 + lane], s1 = v1 * att_src[1 * 32 + lane];
    float s2 = v2 * att_src[2 * 32 + lane], s3 = v3 * att_src[3 * 32 + lane];
    float d0 = v0 * att_dst[0 * 32 + lane], d1 = v1 * att_dst[1 * 32 + lane];
    float d2 = v2 * att_dst[2 * 32 + lane], d3 = v3 * att_dst[3 * 32 + lane];

    #pragma unroll
    for (int m = 16; m >= 1; m >>= 1) {   // wave32 xor reduction
        s0 += __shfl_xor(s0, m, 32); s1 += __shfl_xor(s1, m, 32);
        s2 += __shfl_xor(s2, m, 32); s3 += __shfl_xor(s3, m, 32);
        d0 += __shfl_xor(d0, m, 32); d1 += __shfl_xor(d1, m, 32);
        d2 += __shfl_xor(d2, m, 32); d3 += __shfl_xor(d3, m, 32);
    }

    if (lane < 4) {
        const float s = (lane == 0) ? s0 : (lane == 1) ? s1 : (lane == 2) ? s2 : s3;
        const float d = (lane == 0) ? d0 : (lane == 1) ? d1 : (lane == 2) ? d2 : d3;
        const size_t o = ((size_t)b * HH + lane) * NN + n;    // [B][H][N] layout
        p [o] = __expf(s);         p2[o] = __expf(0.2f * s);
        q [o] = __expf(d);         q2[o] = __expf(0.2f * d);
    }
}

// ------------------------- K3: denominators over i (reference axis=1 norm)
// thread = column j ; coalesced adj[b][i][j] sweep over i
__global__ __launch_bounds__(256) void k_colsum(
    const float* __restrict__ adj,
    const float* __restrict__ p,  const float* __restrict__ p2,
    const float* __restrict__ q,  const float* __restrict__ q2,
    float* __restrict__ invS)
{
    const int j = blockIdx.x * 256 + threadIdx.x;
    const int b = blockIdx.y;

    float qv [HH], q2v[HH], acc[HH];
    #pragma unroll
    for (int h = 0; h < HH; ++h) {
        const size_t o = ((size_t)b * HH + h) * NN + j;
        qv[h] = q[o]; q2v[h] = q2[o]; acc[h] = 0.f;
    }

    const float* __restrict__ ap = adj + (size_t)b * NN * NN + j;
    #pragma unroll 4
    for (int i = 0; i < NN; ++i) {
        float av = ap[(size_t)i * NN];
        if (i == j) av = 1.0f;                 // self-loop
        #pragma unroll
        for (int h = 0; h < HH; ++h) {
            const float pi  = p [((size_t)b * HH + h) * NN + i];  // uniform
            const float p2i = p2[((size_t)b * HH + h) * NN + i];
            const float r   = pi * qv[h];
            const float val = (r > 1.0f) ? r : (p2i * q2v[h]);
            acc[h] += av * val;
        }
    }
    #pragma unroll
    for (int h = 0; h < HH; ++h)
        invS[((size_t)b * HH + h) * NN + j] = 1.0f / acc[h];  // diag term => >0
}

// ----------------------------- K4: aggregation via v_wmma_f32_16x16x32_f16
// block = (b, 32-row i-tile); 8 waves: wave w -> head w>>1, col-half w&1.
// Each wave: 2 WMMAs/chunk (rows M and 16+M) sharing one B operand.
__global__ __launch_bounds__(256) void k_aggregate(
    const float* __restrict__ adj, const _Float16* __restrict__ featT16,
    const float* __restrict__ p,   const float* __restrict__ p2,
    const float* __restrict__ q,   const float* __restrict__ q2,
    const float* __restrict__ invS, const float* __restrict__ bias,
    float* __restrict__ out)
{
    __shared__ float lds_adj[2][ITILE * LPAD];   // double-buffered adj tiles

    const int tid  = threadIdx.x;
    const int b    = blockIdx.x / (NN / ITILE);
    const int i0   = (blockIdx.x % (NN / ITILE)) * ITILE;
    const int wave = tid >> 5;
    const int lane = tid & 31;
    const int h    = wave >> 1;
    const int colb = h * 32 + (wave & 1) * 16;
    const int M    = lane & 15;               // A row / C column
    const int hi   = lane >> 4;

    const size_t hb = (size_t)b * HH + h;
    const float pi0  = p [hb * NN + i0 + M];
    const float p2i0 = p2[hb * NN + i0 + M];
    const float pi1  = p [hb * NN + i0 + 16 + M];
    const float p2i1 = p2[hb * NN + i0 + 16 + M];
    const float* __restrict__ qh  = q    + hb * NN;
    const float* __restrict__ gh  = q2   + hb * NN;
    const float* __restrict__ ish = invS + hb * NN;
    // B operand: lane holds column N=lane&15 ; lanes0-15 K=j0..j0+15,
    // lanes16-31 K=j0+16..j0+31
    const _Float16* __restrict__ bt =
        featT16 + ((size_t)b * HC + colb + M) * NN;

    const float* __restrict__ adjb = adj + (size_t)b * NN * NN;
    // staging geometry: 256 lanes x b128 = 32x32 tile
    const int sii = tid >> 3;                 // row in tile  (0..31)
    const int sjj = (tid & 7) * 4;            // col in tile  (0,4,...,28)
    const float* __restrict__ gsrc0 = adjb + (size_t)(i0 + sii) * NN + sjj;

    v8f acc0 = {}, acc1 = {};

#if defined(HAVE_ASYNC_LDS)
    // ---------------- async double-buffered pipeline (ASYNCcnt-tracked) ----
    async_cp_b128(gsrc0, &lds_adj[0][sii * LPAD + sjj]);       // prologue

    for (int j0 = 0; j0 < NN; j0 += 32) {
        const int sel = (j0 >> 5) & 1;
        if (j0 + 32 < NN) {                                    // stage next
            async_cp_b128(gsrc0 + (j0 + 32),
                          &lds_adj[sel ^ 1][sii * LPAD + sjj]);
            __builtin_amdgcn_s_wait_asynccnt(1);               // cur landed
        } else {
            __builtin_amdgcn_s_wait_asynccnt(0);
        }
        __syncthreads();
        if (j0 == i0) {                       // self-loop diagonal patch
            if (tid < ITILE) lds_adj[sel][tid * LPAD + tid] = 1.0f;
            __syncthreads();
        }
        const float* buf = &lds_adj[sel][0];
#else
    // ---------------------------- synchronous fallback ---------------------
    for (int j0 = 0; j0 < NN; j0 += 32) {
        {
            const int gi = i0 + sii;
            v4f t = *(const v4f*)(gsrc0 + j0);
            #pragma unroll
            for (int e = 0; e < 4; ++e)
                if (gi == j0 + sjj + e) t[e] = 1.0f;           // self-loop
            *(v4f*)&lds_adj[0][sii * LPAD + sjj] = t;
        }
        if (j0 + 32 < NN)
            __builtin_prefetch(gsrc0 + (j0 + 32), 0, 1);
        __syncthreads();
        const float* buf = &lds_adj[0][0];
#endif
        // A operands 16x32 f16: lanes0-15 K{0..15},
        // lanes16-31 K{16..31}; on-the-fly alpha, no exp inner loop.
        v16h A0, A1;
        #pragma unroll
        for (int r2 = 0; r2 < 2; ++r2) {
            const int kb = hi * 16 + r2 * 8;
            const v4f qa = *(const v4f*)(qh  + j0 + kb);
            const v4f qb = *(const v4f*)(qh  + j0 + kb + 4);
            const v4f ga = *(const v4f*)(gh  + j0 + kb);
            const v4f gb = *(const v4f*)(gh  + j0 + kb + 4);
            const v4f sa = *(const v4f*)(ish + j0 + kb);
            const v4f sb = *(const v4f*)(ish + j0 + kb + 4);
            const v4f x0 = *(const v4f*)&buf[M * LPAD + kb];
            const v4f x1 = *(const v4f*)&buf[M * LPAD + kb + 4];
            const v4f y0 = *(const v4f*)&buf[(16 + M) * LPAD + kb];
            const v4f y1 = *(const v4f*)&buf[(16 + M) * LPAD + kb + 4];
            #pragma unroll
            for (int t = 0; t < 4; ++t) {
                const float ra  = pi0 * qa[t];
                const float va  = (ra  > 1.0f) ? ra  : (p2i0 * ga[t]);
                const float rb  = pi0 * qb[t];
                const float vb  = (rb  > 1.0f) ? rb  : (p2i0 * gb[t]);
                A0[r2 * 8 + t]     = (_Float16)(x0[t] * va * sa[t]);
                A0[r2 * 8 + 4 + t] = (_Float16)(x1[t] * vb * sb[t]);
                const float rc  = pi1 * qa[t];
                const float vc  = (rc  > 1.0f) ? rc  : (p2i1 * ga[t]);
                const float rd  = pi1 * qb[t];
                const float vd  = (rd  > 1.0f) ? rd  : (p2i1 * gb[t]);
                A1[r2 * 8 + t]     = (_Float16)(y0[t] * vc * sa[t]);
                A1[r2 * 8 + 4 + t] = (_Float16)(y1[t] * vd * sb[t]);
            }
        }

        const v16h Bv = *(const v16h*)(bt + j0 + hi * 16);  // 2x b128, shared

        acc0 = __builtin_amdgcn_wmma_f32_16x16x32_f16(
            false, A0, false, Bv, (short)0, acc0, false, false);
        acc1 = __builtin_amdgcn_wmma_f32_16x16x32_f16(
            false, A1, false, Bv, (short)0, acc1, false, false);
        __syncthreads();
    }

    const int cg = colb + M;
    #pragma unroll
    for (int r = 0; r < 8; ++r) {
        const int row0 = i0 + r + hi * 8;
        out[((size_t)b * NN + row0) * HC + cg]      = acc0[r] + bias[cg];
        out[((size_t)b * NN + row0 + 16) * HC + cg] = acc1[r] + bias[cg];
    }
}

// ---------------------------------------------------------------------------
extern "C" void kernel_launch(void* const* d_in, const int* in_sizes, int n_in,
                              void* d_out, int out_size, void* d_ws, size_t ws_size,
                              hipStream_t stream)
{
    const float* x       = (const float*)d_in[0];
    const float* adj     = (const float*)d_in[1];
    const float* W       = (const float*)d_in[2];
    const float* att_src = (const float*)d_in[3];
    const float* att_dst = (const float*)d_in[4];
    const float* bias    = (const float*)d_in[5];
    float*       out     = (float*)d_out;

    char* ws = (char*)d_ws;
    float*    feat    = (float*)   (ws + 0);                 // 4 MB
    _Float16* featT16 = (_Float16*)(ws + 4194304);           // 2 MB
    float*    p       = (float*)   (ws + 6291456);           // 128 KB each
    float*    p2      = (float*)   (ws + 6422528);
    float*    q       = (float*)   (ws + 6553600);
    float*    q2      = (float*)   (ws + 6684672);
    float*    invS    = (float*)   (ws + 6815744);

    k_project  <<<BB * NN / 16, 256, 0, stream>>>(x, W, feat, featT16);
    k_attn     <<<BB * NN / 8,  256, 0, stream>>>(feat, att_src, att_dst, p, p2, q, q2);
    k_colsum   <<<dim3(NN / 256, BB), 256, 0, stream>>>(adj, p, p2, q, q2, invS);
    k_aggregate<<<BB * NN / ITILE, 256, 0, stream>>>(adj, featT16, p, p2, q, q2,
                                                     invS, bias, out);
}
